// OptimizedMoBAAttention_79087527788600
// MI455X (gfx1250) — compile-verified, matrix operand order checked by reading, not run on hardware
//
#include <hip/hip_runtime.h>
#include <stdint.h>

// ---------------- CDNA5 wave32 WMMA types ----------------
typedef __attribute__((ext_vector_type(16))) __bf16 v16bf;
typedef __attribute__((ext_vector_type(8)))  float  v8f;

union ABf { v16bf bf; uint4 u[2]; };

#define BD   2
#define SD   2048
#define DM   1024
#define HN   16
#define HD   64
#define BLK  256
#define NBLK 8
#define MTOT (BD * SD)   // 4096

// ---- gfx1250 async global->LDS path (probe-guarded, confirmed present) ----
#if defined(__has_builtin)
#if __has_builtin(__builtin_amdgcn_global_load_async_to_lds_b128) && \
    __has_builtin(__builtin_amdgcn_s_wait_asynccnt)
#define HAVE_ASYNC_LDS 1
#endif
#endif

#ifdef HAVE_ASYNC_LDS
typedef int v4i_gcc __attribute__((vector_size(16)));
typedef __attribute__((address_space(1))) v4i_gcc* g_v4i_ptr;
typedef __attribute__((address_space(3))) v4i_gcc* l_v4i_ptr;
__device__ __forceinline__ void async_copy16B(const unsigned short* g, unsigned short* l) {
  __builtin_amdgcn_global_load_async_to_lds_b128(
      (g_v4i_ptr)(v4i_gcc*)(void*)const_cast<unsigned short*>(g),
      (l_v4i_ptr)(v4i_gcc*)(void*)l, 0, 0);
}
#endif

__device__ __forceinline__ unsigned wave_ballot32(bool p) {
#if defined(__has_builtin)
#if __has_builtin(__builtin_amdgcn_ballot_w32)
  return __builtin_amdgcn_ballot_w32(p);
#else
  return (unsigned)__ballot((int)p);
#endif
#else
  return (unsigned)__ballot((int)p);
#endif
}

__device__ __forceinline__ unsigned short f2bf(float f) {
  unsigned int u = __float_as_uint(f);
  u += 0x7FFFu + ((u >> 16) & 1u);              // round-to-nearest-even
  return (unsigned short)(u >> 16);
}
__device__ __forceinline__ float bf2f(unsigned short h) {
  return __uint_as_float(((unsigned int)h) << 16);
}

__device__ __forceinline__ v8f wmma_bf16(const ABf& a, const ABf& b, v8f c) {
  return __builtin_amdgcn_wmma_f32_16x16x32_bf16(false, a.bf, false, b.bf,
                                                 (short)0, c, false, false);
}

// wave-local LDS producer->consumer ordering (regions are per-wave private,
// wave32 executes in lockstep; only need DScnt drained + a scheduling fence)
__device__ __forceinline__ void wave_sync_lds() {
  __builtin_amdgcn_wave_barrier();
  asm volatile("s_wait_dscnt 0x0" ::: "memory");
  __builtin_amdgcn_wave_barrier();
}

// ---------------- fp32 -> bf16 conversion ----------------
__global__ __launch_bounds__(256) void conv_f2bf_kernel(
    const float* __restrict__ src, unsigned short* __restrict__ dst, int n) {
  int i = blockIdx.x * 256 + threadIdx.x;
  if (i < n) dst[i] = f2bf(src[i]);
}

// transpose + convert a [K=1024][N=1024] weight to bf16 [N][K]
__global__ __launch_bounds__(256) void conv_wT_kernel(
    const float* __restrict__ w, unsigned short* __restrict__ wt, int n) {
  int i = blockIdx.x * 256 + threadIdx.x;
  if (i < n) {
    int nn = i >> 10;
    int kk = i & 1023;
    wt[i] = f2bf(w[(size_t)kk * DM + nn]);
  }
}

// ---------------- bf16 WMMA GEMM : C = A[M x 1024] * Bt^T + bias ----------------
// Block tile 256(M) x 64(N), 8 waves (wave = 32 rows), K-step 64 -> 16 WMMA/iter.
// Double-buffered async global->LDS staging of the B tile: issue copy for
// tile t+1, compute tile t, wait ASYNCcnt only at the top of the next iter.
// mode 0: out bf16 per-head layout  [B,H,S,hd]
// mode 1: out bf16 head-transposed  [B,H,hd,S]   (for V)
// mode 2: out fp32 plain            [M][N]       (final projection)
__global__ __launch_bounds__(256) void gemm_bf16_kernel(
    const unsigned short* __restrict__ A, const unsigned short* __restrict__ Bt,
    const float* __restrict__ bias, void* __restrict__ Out, int mode) {
  __shared__ __align__(16) unsigned short sB[2 * 64 * 64];   // 2 x 8 KB B tiles
  const int K = DM;
  const int lane = threadIdx.x & 31;
  const int wave = threadIdx.x >> 5;
  const int m0 = blockIdx.x * 256 + wave * 32;
  const int n0 = blockIdx.y * 64;
  const int koff = (lane >> 4) << 3;            // 16-bit frag: lanes 16-31 hold K+8
  const int arow0 = m0 + (lane & 15);
  v8f acc[2][4] = {};

  // staging assignment for this thread: row t>>2 (0..63), 16 elems at (t&3)*16
  const int srow = threadIdx.x >> 2;
  const int scol = (threadIdx.x & 3) << 4;

  auto stage = [&](int k0, int buf) {
    const unsigned short* src = &Bt[(size_t)(n0 + srow) * K + k0 + scol];
    unsigned short* dstl = &sB[buf * 64 * 64 + srow * 64 + scol];
#ifdef HAVE_ASYNC_LDS
    async_copy16B(src, dstl);
    async_copy16B(src + 8, dstl + 8);
#else
    *(uint4*)(dstl)     = *(const uint4*)(src);
    *(uint4*)(dstl + 8) = *(const uint4*)(src + 8);
#endif
  };

  const int NT = K / 64;                        // 16 K-tiles
  stage(0, 0);
  for (int t = 0; t < NT; ++t) {
#ifdef HAVE_ASYNC_LDS
    __builtin_amdgcn_s_wait_asynccnt(0);        // staged tile t has landed
#endif
    __syncthreads();                            // visible to all waves
    if (t + 1 < NT) stage((t + 1) * 64, (t + 1) & 1);   // overlap next copy

    const int k0 = t * 64;
    const unsigned short* sBt = &sB[(t & 1) * 64 * 64];
#pragma unroll
    for (int kc = 0; kc < 2; ++kc) {
      ABf a[2];
#pragma unroll
      for (int rt = 0; rt < 2; ++rt) {
        const unsigned short* ap = &A[(size_t)(arow0 + rt * 16) * K + k0 + kc * 32 + koff];
        a[rt].u[0] = *(const uint4*)(ap);
        a[rt].u[1] = *(const uint4*)(ap + 16);
        if (kc == 0 && t + 1 < NT) __builtin_prefetch(ap + 64, 0, 1);
      }
#pragma unroll
      for (int nt = 0; nt < 4; ++nt) {
        ABf b;
        const unsigned short* bp = &sBt[(nt * 16 + (lane & 15)) * 64 + kc * 32 + koff];
        b.u[0] = *(const uint4*)(bp);
        b.u[1] = *(const uint4*)(bp + 16);
#pragma unroll
        for (int rt = 0; rt < 2; ++rt)
          acc[rt][nt] = wmma_bf16(a[rt], b, acc[rt][nt]);
      }
    }
    __syncthreads();                            // retire reads before re-stage
  }

  // epilogue: C frag — lanes 0-15: N=col, M=vgpr; lanes 16-31: M=vgpr+8
#pragma unroll
  for (int rt = 0; rt < 2; ++rt) {
#pragma unroll
    for (int nt = 0; nt < 4; ++nt) {
      int col = n0 + nt * 16 + (lane & 15);
      float bv = bias[col];
#pragma unroll
      for (int r = 0; r < 8; ++r) {
        int m = m0 + rt * 16 + r + ((lane >> 4) << 3);
        float val = acc[rt][nt][r] + bv;
        int b_ = m >> 11, s = m & 2047;
        int h = col >> 6, d = col & 63;
        if (mode == 0) {
          ((unsigned short*)Out)[(((size_t)(b_ * HN + h) * SD) + s) * HD + d] = f2bf(val);
        } else if (mode == 1) {
          ((unsigned short*)Out)[(((size_t)(b_ * HN + h) * HD) + d) * SD + s] = f2bf(val);
        } else {
          ((float*)Out)[(size_t)m * DM + col] = val;
        }
      }
    }
  }
}

// ---------------- per-block key mean ----------------
__global__ __launch_bounds__(256) void kmean_kernel(
    const unsigned short* __restrict__ kb, float* __restrict__ km) {
  int t = blockIdx.x * 256 + threadIdx.x;   // B*H*NB*HD = 16384
  int d = t & 63;
  int j = (t >> 6) & 7;
  int bh = t >> 9;
  const unsigned short* p = &kb[(((size_t)bh * SD) + j * BLK) * HD + d];
  float acc = 0.f;
  for (int s = 0; s < BLK; ++s) acc += bf2f(p[(size_t)s * HD]);
  km[t] = acc * (1.f / BLK);
}

// ---------------- gating: top-3 previous-block selection ----------------
__global__ __launch_bounds__(256) void gate_kernel(
    const unsigned short* __restrict__ qb, const float* __restrict__ km,
    float* __restrict__ sel) {
  int t = blockIdx.x * 256 + threadIdx.x;   // B*H*S = 65536
  int s = t & 2047;
  int bh = t >> 11;
  int qblk = s >> 8;
  const unsigned short* qr = &qb[((size_t)bh * SD + s) * HD];
  float g[8];
#pragma unroll
  for (int j = 0; j < 8; ++j) {
    if (j <= qblk) {
      const float* kmr = &km[((size_t)bh * 8 + j) * HD];
      float acc = 0.f;
      for (int d = 0; d < HD; ++d) acc += bf2f(qr[d]) * kmr[d];
      g[j] = acc;                           // gate has NO 1/sqrt(hd) scale (as in source)
    } else {
      g[j] = -3.0e38f;
    }
  }
  unsigned mask = 0;
  for (int it = 0; it < 3; ++it) {
    float best = -3.0e38f; int bj = -1;
#pragma unroll
    for (int j = 0; j < 8; ++j)
      if (!((mask >> j) & 1u) && g[j] > best) { best = g[j]; bj = j; }
    if (bj >= 0) mask |= 1u << bj;
  }
  float* sr = &sel[(size_t)t * 8];
#pragma unroll
  for (int j = 0; j < 8; ++j)
    sr[j] = (((mask >> j) & 1u) && (j < qblk)) ? 1.0f : 0.0f;
}

// ---------------- MoBA attention core ----------------
// grid: B*H*NBLK*2 blocks, 128 threads (4 waves), wave = 32 queries.
// All LDS regions are per-wave private -> no block barriers; waves skip
// previous blocks none of their 32 rows selected (ballot).
__global__ __launch_bounds__(128) void moba_attn_kernel(
    const unsigned short* __restrict__ qb, const unsigned short* __restrict__ kb,
    const unsigned short* __restrict__ vtb, const float* __restrict__ sel,
    unsigned short* __restrict__ attnb) {
  extern __shared__ char dynsm[];
  float* sc = (float*)dynsm;                                     // 4*32*256 f32
  unsigned short* pb = (unsigned short*)(dynsm + 4 * 32 * 256 * sizeof(float));
  const float scale = 0.125f;                                    // 1/sqrt(64)

  const int lane = threadIdx.x & 31;
  const int wave = threadIdx.x >> 5;
  const int blk = blockIdx.x;
  const int hh = blk & 1;
  const int jq = (blk >> 1) & 7;
  const int h  = (blk >> 4) & 15;
  const int b_ = blk >> 8;
  const int bh = b_ * HN + h;
  const int qw0 = hh * 128 + wave * 32;
  float* scw = sc + (size_t)wave * 32 * 256;
  unsigned short* pbw = pb + (size_t)wave * 32 * 256;
  const int koff = (lane >> 4) << 3;

  // Q fragments: 2 row-tiles x 2 K-chunks (hd=64)
  ABf qf[2][2];
#pragma unroll
  for (int rt = 0; rt < 2; ++rt) {
    int qrow = jq * BLK + qw0 + rt * 16 + (lane & 15);
    const unsigned short* qp = &qb[((size_t)bh * SD + qrow) * HD];
#pragma unroll
    for (int kc = 0; kc < 2; ++kc) {
      qf[rt][kc].u[0] = *(const uint4*)(qp + kc * 32 + koff);
      qf[rt][kc].u[1] = *(const uint4*)(qp + kc * 32 + koff + 16);
    }
  }

  v8f oacc[2][4] = {};
  const int myrow = jq * BLK + qw0 + lane;
  const float* selr = &sel[((size_t)bh * SD + myrow) * 8];

  for (int j = 0; j <= jq; ++j) {
    // per-lane gate for this lane's softmax row; wave skips dead blocks
    float gatef = (j == jq) ? 1.0f : selr[j];
    if (wave_ballot32(gatef != 0.0f) == 0u) continue;

    // ---- phase 1: S = scale * Q K^T ----
#pragma unroll 1
    for (int nt = 0; nt < 16; ++nt) {
      ABf kf[2];
      int key = j * BLK + nt * 16 + (lane & 15);
      const unsigned short* kp = &kb[((size_t)bh * SD + key) * HD];
#pragma unroll
      for (int kc = 0; kc < 2; ++kc) {
        kf[kc].u[0] = *(const uint4*)(kp + kc * 32 + koff);
        kf[kc].u[1] = *(const uint4*)(kp + kc * 32 + koff + 16);
      }
#pragma unroll
      for (int rt = 0; rt < 2; ++rt) {
        v8f sfrag = {};
        sfrag = wmma_bf16(qf[rt][0], kf[0], sfrag);
        sfrag = wmma_bf16(qf[rt][1], kf[1], sfrag);
        int kl = nt * 16 + (lane & 15);
#pragma unroll
        for (int r = 0; r < 8; ++r) {
          int rl = rt * 16 + r + ((lane >> 4) << 3);
          float v = sfrag[r] * scale;
          if (j == jq) {                              // self block: causal
            int qpos = qw0 + rl;
            if (kl > qpos) v = -3.0e38f;
          }
          scw[(size_t)rl * 256 + kl] = v;
        }
      }
    }
    wave_sync_lds();

    // ---- phase 2: per-row softmax (single exp pass), fold sel/sum into P ----
    {
      float4* sp = (float4*)(scw + (size_t)lane * 256);
      float m = -3.0e38f;
      for (int i = 0; i < 64; ++i) {
        float4 v = sp[i];
        m = fmaxf(m, fmaxf(fmaxf(v.x, v.y), fmaxf(v.z, v.w)));
      }
      float sum = 0.f;
      for (int i = 0; i < 64; ++i) {
        float4 v = sp[i];
        float4 e;
        e.x = __expf(v.x - m); e.y = __expf(v.y - m);
        e.z = __expf(v.z - m); e.w = __expf(v.w - m);
        sum += (e.x + e.y) + (e.z + e.w);
        sp[i] = e;                                   // write e back (exp once)
      }
      float f = gatef / sum;
      unsigned int* pr = (unsigned int*)(pbw + (size_t)lane * 256);
      const float2* sp2 = (const float2*)(scw + (size_t)lane * 256);
      for (int i = 0; i < 128; ++i) {
        float2 v = sp2[i];
        unsigned lo = f2bf(v.x * f);
        unsigned hi = f2bf(v.y * f);
        pr[i] = lo | (hi << 16);
      }
    }
    wave_sync_lds();

    // ---- phase 3: O += P V ----
#pragma unroll 1
    for (int kc = 0; kc < 8; ++kc) {
      ABf pa[2];
#pragma unroll
      for (int rt = 0; rt < 2; ++rt) {
        const unsigned short* pp =
            &pbw[(size_t)(rt * 16 + (lane & 15)) * 256 + kc * 32 + koff];
        pa[rt].u[0] = *(const uint4*)(pp);
        pa[rt].u[1] = *(const uint4*)(pp + 16);
      }
#pragma unroll
      for (int dt = 0; dt < 4; ++dt) {
        ABf vf;
        int d = dt * 16 + (lane & 15);
        const unsigned short* vp =
            &vtb[((size_t)bh * HD + d) * SD + j * BLK + kc * 32 + koff];
        vf.u[0] = *(const uint4*)(vp);
        vf.u[1] = *(const uint4*)(vp + 16);
#pragma unroll
        for (int rt = 0; rt < 2; ++rt)
          oacc[rt][dt] = wmma_bf16(pa[rt], vf, oacc[rt][dt]);
      }
    }
    wave_sync_lds();
  }

  // epilogue: bf16 attn buffer [B,S,D], D index = h*64 + d
#pragma unroll
  for (int rt = 0; rt < 2; ++rt) {
#pragma unroll
    for (int dt = 0; dt < 4; ++dt) {
      int col = h * HD + dt * 16 + (lane & 15);
#pragma unroll
      for (int r = 0; r < 8; ++r) {
        int srow = jq * BLK + qw0 + rt * 16 + r + ((lane >> 4) << 3);
        attnb[((size_t)b_ * SD + srow) * DM + col] = f2bf(oacc[rt][dt][r]);
      }
    }
  }
}

// ---------------- host-side orchestration ----------------
extern "C" void kernel_launch(void* const* d_in, const int* in_sizes, int n_in,
                              void* d_out, int out_size, void* d_ws, size_t ws_size,
                              hipStream_t stream) {
  (void)in_sizes; (void)n_in; (void)out_size; (void)ws_size;
  const float* x     = (const float*)d_in[0];
  const float* qw    = (const float*)d_in[1];
  const float* qbias = (const float*)d_in[2];
  const float* kw    = (const float*)d_in[3];
  const float* kbias = (const float*)d_in[4];
  const float* vw    = (const float*)d_in[5];
  const float* vbias = (const float*)d_in[6];
  const float* ow    = (const float*)d_in[7];
  const float* obias = (const float*)d_in[8];

  char* ws = (char*)d_ws;
  size_t off = 0;
  auto alloc = [&](size_t bytes) -> void* {
    void* p = ws + off;
    off += (bytes + 255) & ~(size_t)255;
    return p;
  };
  unsigned short* xb    = (unsigned short*)alloc((size_t)MTOT * DM * 2);
  unsigned short* wqT   = (unsigned short*)alloc((size_t)DM * DM * 2);
  unsigned short* wkT   = (unsigned short*)alloc((size_t)DM * DM * 2);
  unsigned short* wvT   = (unsigned short*)alloc((size_t)DM * DM * 2);
  unsigned short* woT   = (unsigned short*)alloc((size_t)DM * DM * 2);
  unsigned short* qbuf  = (unsigned short*)alloc((size_t)MTOT * DM * 2);  // [B,H,S,hd]
  unsigned short* kbuf  = (unsigned short*)alloc((size_t)MTOT * DM * 2);  // [B,H,S,hd]
  unsigned short* vtbuf = (unsigned short*)alloc((size_t)MTOT * DM * 2);  // [B,H,hd,S]
  unsigned short* attnb = (unsigned short*)alloc((size_t)MTOT * DM * 2);  // [B,S,D]
  float* km             = (float*)alloc((size_t)BD * HN * NBLK * HD * 4);
  float* selbuf         = (float*)alloc((size_t)BD * HN * SD * NBLK * 4);

  conv_f2bf_kernel<<<(MTOT * DM) / 256, 256, 0, stream>>>(x, xb, MTOT * DM);
  conv_wT_kernel<<<(DM * DM) / 256, 256, 0, stream>>>(qw, wqT, DM * DM);
  conv_wT_kernel<<<(DM * DM) / 256, 256, 0, stream>>>(kw, wkT, DM * DM);
  conv_wT_kernel<<<(DM * DM) / 256, 256, 0, stream>>>(vw, wvT, DM * DM);
  conv_wT_kernel<<<(DM * DM) / 256, 256, 0, stream>>>(ow, woT, DM * DM);

  dim3 gg(MTOT / 256, DM / 64);
  gemm_bf16_kernel<<<gg, 256, 0, stream>>>(xb, wqT, qbias, qbuf, 0);
  gemm_bf16_kernel<<<gg, 256, 0, stream>>>(xb, wkT, kbias, kbuf, 0);
  gemm_bf16_kernel<<<gg, 256, 0, stream>>>(xb, wvT, vbias, vtbuf, 1);

  kmean_kernel<<<(BD * HN * NBLK * HD) / 256, 256, 0, stream>>>(kbuf, km);
  gate_kernel<<<(BD * HN * SD) / 256, 256, 0, stream>>>(qbuf, km, selbuf);

  moba_attn_kernel<<<BD * HN * NBLK * 2, 128, 4 * 32 * 256 * 6, stream>>>(
      qbuf, kbuf, vtbuf, selbuf, attnb);

  gemm_bf16_kernel<<<gg, 256, 0, stream>>>(attnb, woT, obias, (float*)d_out, 2);
}
